// Attention_15994458210542
// MI455X (gfx1250) — matrix operand order repaired and checked by reading, not compiled, not run on hardware
//
#include <hip/hip_runtime.h>

// ---------------------------------------------------------------------------
// MI455X (gfx1250) factorized-rotary attention, bf16 WMMA pipeline.
//   K1: qkv projection GEMM (v_wmma_f32_16x16x32_bf16)
//   K2: flash attention, 128 problems x 16 superblocks, 4 waves/WG
//   K3: output projection GEMM
// All WMMA fragments are contiguous 16B-aligned LDS runs -> ds_load_b128.
// bf16 conversion goes through the compiler's native __bf16 lowering.
// Workspace: q_base | k_base | v_base | o_mean  (4 x 1MB f32)
// ---------------------------------------------------------------------------

typedef __bf16 bf16_t;
typedef __attribute__((ext_vector_type(2)))  __bf16 v2bf;
typedef __attribute__((ext_vector_type(16))) __bf16 v16bf;
typedef __attribute__((ext_vector_type(8)))  float  v8f;

union ABf { v16bf v; unsigned u[8]; uint4 q[2]; };

#define ASTR 40      // A/P tile row stride in u16 (80B rows, 16B aligned)
#define ASTR32 20    // same in u32
#define BPSTR 20     // GEMM B tile: u32 per column (pair-major)
#define KSTR 72      // attention K tile row stride in u16 (144B rows)
#define KSTR32 36
#define LOG2E  1.4426950408889634f
#define NEGINF -3.0e38f

// f32 -> bf16 via the compiler's native lowering for gfx1250.
static __device__ inline unsigned short f2bf(float f) {
    __bf16 b = (__bf16)f;
    unsigned short out;
    __builtin_memcpy(&out, &b, 2);
    return out;
}
// 2xf32 -> packed 2xbf16.
static __device__ inline unsigned packf2(float lo, float hi) {
#if __has_builtin(__builtin_amdgcn_cvt_pk_bf16_f32)
    auto r = __builtin_amdgcn_cvt_pk_bf16_f32(lo, hi);
    static_assert(sizeof(r) == 4, "unexpected cvt_pk_bf16_f32 result size");
    unsigned out;
    __builtin_memcpy(&out, &r, 4);
    return out;
#else
    v2bf t;
    t[0] = (__bf16)lo;
    t[1] = (__bf16)hi;
    unsigned out;
    __builtin_memcpy(&out, &t, 4);
    return out;
#endif
}
static __device__ inline v8f wmma_bf16(v16bf a, v16bf b, v8f c) {
    return __builtin_amdgcn_wmma_f32_16x16x32_bf16(false, a, false, b,
                                                   (short)0, c, false, false);
}
static __device__ inline v8f zero8() {
    v8f z;
#pragma unroll
    for (int r = 0; r < 8; ++r) z[r] = 0.0f;
    return z;
}
// 8 contiguous u32 (16B-aligned): attention K frags.
static __device__ inline v16bf load_frag_contig(const unsigned* p) {
    ABf a;
    a.q[0] = *(const uint4*)p;
    a.q[1] = *(const uint4*)(p + 4);
    return a.v;
}
// A-layout frag: two contiguous 4-u32 runs with an 8-u32 gap.
static __device__ inline v16bf load_frag_split(const unsigned* p) {
    ABf a;
    a.q[0] = *(const uint4*)p;
    a.q[1] = *(const uint4*)(p + 8);
    return a.v;
}

// ---------------------------------------------------------------------------
// Kernel 1: q/k/v projection.  C[row,col] = x[row,:] . W[col,:] + b[col]
// 192 blocks: z(3) x rt(8) x ct(8).  Output in [h][s2][l=s1][d] layout.
// ---------------------------------------------------------------------------
__global__ __launch_bounds__(128) void qkv_gemm_kernel(
    const float* __restrict__ x,
    const float* __restrict__ qw, const float* __restrict__ qbias,
    const float* __restrict__ kw, const float* __restrict__ kbias,
    const float* __restrict__ vw, const float* __restrict__ vbias,
    float* __restrict__ qo, float* __restrict__ ko, float* __restrict__ vo)
{
    __shared__ __align__(16) unsigned aT[64 * ASTR32];   // A rows, u16-pairs packed
    __shared__ __align__(16) unsigned bP[64 * BPSTR];    // B pair-major [c][kpair]
    const int tid  = threadIdx.x;
    const int lane = tid & 31, w = tid >> 5;
    const int half = lane >> 4, m = lane & 15;
    const int t = blockIdx.x;
    const int z = t / 64, rem = t % 64;
    const int rt = rem >> 3, ct = rem & 7;
    const float* W  = (z == 0) ? qw : (z == 1) ? kw : vw;
    const float* Bv = (z == 0) ? qbias : (z == 1) ? kbias : vbias;
    float* Out      = (z == 0) ? qo : (z == 1) ? ko : vo;

    v8f acc[4];
#pragma unroll
    for (int cg = 0; cg < 4; ++cg) acc[cg] = zero8();

    for (int kk = 0; kk < 512; kk += 32) {
        __syncthreads();
#pragma unroll
        for (int e = 0; e < 8; ++e) {           // A chunk 64x32 as 1024 pairs
            int idx = e * 128 + tid;
            int r = idx >> 4, cp = idx & 15;
            const float* xr = &x[(rt * 64 + r) * 512 + kk + 2 * cp];
            aT[r * ASTR32 + cp] = packf2(xr[0], xr[1]);
        }
#pragma unroll
        for (int e = 0; e < 8; ++e) {           // B chunk (W^T) pair-major
            int idx = e * 128 + tid;
            int kp = idx & 15, c = idx >> 4;
            const float* wr = &W[(ct * 64 + c) * 512 + kk + 2 * kp];
            bP[c * BPSTR + kp] = packf2(wr[0], wr[1]);
        }
        __syncthreads();

        v16bf a = load_frag_split(&aT[(w * 16 + m) * ASTR32 + half * 4]);
#pragma unroll
        for (int cg = 0; cg < 4; ++cg) {
            v16bf b = load_frag_split(&bP[(cg * 16 + m) * BPSTR + half * 4]);
            acc[cg] = wmma_bf16(a, b, acc[cg]);
        }
    }
#pragma unroll
    for (int cg = 0; cg < 4; ++cg)
#pragma unroll
        for (int r = 0; r < 8; ++r) {
            int row = rt * 64 + w * 16 + r + 8 * half;   // = s1*16+s2
            int col = ct * 64 + cg * 16 + m;             // = h*64+d
            int h = col >> 6, d = col & 63, s1 = row >> 4, s2 = row & 15;
            Out[((h * 16 + s2) * 32 + s1) * 64 + d] = acc[cg][r] + Bv[col];
        }
}

// ---------------------------------------------------------------------------
// Kernel 2: flash attention over LL=1024 factorized positions.
// grid = 128 problems * 16 superblocks; block = 128 thr (4 waves).
// ---------------------------------------------------------------------------
__global__ __launch_bounds__(128) void attn_kernel(
    const float* __restrict__ qB, const float* __restrict__ kB,
    const float* __restrict__ vB,
    const float* __restrict__ cosT, const float* __restrict__ sinT,
    float* __restrict__ oM)
{
    __shared__ __align__(16) float          kbS[32 * 64];       // base K (f32)
    __shared__ __align__(16) unsigned short kT[32 * KSTR];      // rotated K bf16
    __shared__ __align__(16) unsigned short pT[4 * 16 * ASTR];  // per-wave P

    const int tid  = threadIdx.x;
    const int lane = tid & 31, w = tid >> 5;
    const int half = lane >> 4, n = lane & 15;
    const int prob = blockIdx.x >> 4;                  // h*16 + s2
    const int sb   = blockIdx.x & 15;

    const float* qp = qB + prob * 2048;
    const float* kp = kB + prob * 2048;
    const float* vp = vB + prob * 2048;
    float*       op = oM + prob * 2048;

#pragma unroll
    for (int e = 0; e < 16; ++e) {                     // stage base K tile
        int idx = e * 128 + tid;
        kbS[idx] = kp[idx];
    }

    const int qt0 = sb * 64 + w * 16;                  // this wave's q tile
    const int iq  = qt0 + n;                           // A-frag row index

    // Rotated, pre-scaled Q fragments built straight into registers.
    ABf qa0, qa1;
    {
        const float* qr = qp + (iq >> 5) * 64;
        const float* cr = cosT + iq * 64;
        const float* sr = sinT + iq * 64;
#pragma unroll
        for (int v = 0; v < 8; ++v) {
            int d0 = half * 16 + 2 * v;  // 0..31
            float lo[2], hi[2];
#pragma unroll
            for (int u = 0; u < 2; ++u) {
                int d = d0 + u;
                lo[u] = 0.125f * (qr[d] * cr[d] - qr[d + 32] * sr[d]);
                hi[u] = 0.125f * (qr[d + 32] * cr[d + 32] + qr[d] * sr[d + 32]);
            }
            qa0.u[v] = packf2(lo[0], lo[1]);
            qa1.u[v] = packf2(hi[0], hi[1]);
        }
    }

    // Loop-invariant V B-fragments (keys repeat mod 32, no rotary on V).
    ABf vf[4];
#pragma unroll
    for (int cg = 0; cg < 4; ++cg)
#pragma unroll
        for (int v = 0; v < 8; ++v) {
            int k0 = (v & 3) * 2 + ((v >> 2) << 4) + half * 8;
            vf[cg].u[v] = packf2(vp[k0 * 64 + cg * 16 + n],
                                 vp[(k0 + 1) * 64 + cg * 16 + n]);
        }

    v8f o0 = zero8(), o1 = zero8(), o2 = zero8(), o3 = zero8();
    float rmax[8], rsum[8];
#pragma unroll
    for (int r = 0; r < 8; ++r) { rmax[r] = NEGINF; rsum[r] = 0.0f; }

    const int myJmax = ((qt0 + 15) >> 5) << 5;
    const int sbJmax = ((sb * 64 + 63) >> 5) << 5;
    unsigned short* pW = &pT[w * 16 * ASTR];
    const unsigned* pW32 = (const unsigned*)pW;
    unsigned* kT32 = (unsigned*)kT;

    for (int j0 = 0; j0 <= sbJmax; j0 += 32) {
        __syncthreads();
        // Cooperatively build rotated K tile: 1024 element-pairs, one packed
        // u32 DS store per pair (d even: both halves share the rot branch).
#pragma unroll
        for (int e = 0; e < 8; ++e) {
            int idx = e * 128 + tid;
            int r = idx >> 5, dp = idx & 31;
            int d = 2 * dp;
            int j = j0 + r;
            const float* cj = &cosT[j * 64 + d];
            const float* sj = &sinT[j * 64 + d];
            float b0 = kbS[r * 64 + d], b1 = kbS[r * 64 + d + 1];
            float r0, r1;
            if (d < 32) { r0 = -kbS[r * 64 + d + 32]; r1 = -kbS[r * 64 + d + 33]; }
            else        { r0 =  kbS[r * 64 + d - 32]; r1 =  kbS[r * 64 + d - 31]; }
            kT32[r * KSTR32 + dp] = packf2(b0 * cj[0] + r0 * sj[0],
                                           b1 * cj[1] + r1 * sj[1]);
        }
        if (tid == 0 && j0 + 32 <= sbJmax) {           // prefetch next rotary block
            __builtin_prefetch(&cosT[(j0 + 32) * 64], 0, 1);
            __builtin_prefetch(&sinT[(j0 + 32) * 64], 0, 1);
        }
        __syncthreads();
        if (j0 > myJmax) continue;                     // wave done (uniform)

        // QK^T:  S[16x32] = Q(16x64) . K_rot^T(64x32); fragments are
        // contiguous 16-u16 runs -> ds_load_b128 pairs.
        v16bf b0a = load_frag_contig((const unsigned*)&kT[n * KSTR + half * 16]);
        v16bf b0b = load_frag_contig((const unsigned*)&kT[n * KSTR + 32 + half * 16]);
        v16bf b1a = load_frag_contig((const unsigned*)&kT[(16 + n) * KSTR + half * 16]);
        v16bf b1b = load_frag_contig((const unsigned*)&kT[(16 + n) * KSTR + 32 + half * 16]);

        v8f s0 = zero8(), s1 = zero8();
        s0 = wmma_bf16(qa0.v, b0a, s0);
        s0 = wmma_bf16(qa1.v, b0b, s0);
        s1 = wmma_bf16(qa0.v, b1a, s1);
        s1 = wmma_bf16(qa1.v, b1b, s1);

        // Causal mask + online softmax (per-row stats live in the 16-lane half).
        float t[8];
#pragma unroll
        for (int r = 0; r < 8; ++r) {
            int im  = qt0 + r + 8 * half;
            if (j0 + n > im)      s0[r] = NEGINF;
            if (j0 + 16 + n > im) s1[r] = NEGINF;
            t[r] = fmaxf(s0[r], s1[r]);
        }
#pragma unroll
        for (int off = 1; off < 16; off <<= 1)
#pragma unroll
            for (int r = 0; r < 8; ++r)
                t[r] = fmaxf(t[r], __shfl_xor(t[r], off, 16));
        float scl[8];
#pragma unroll
        for (int r = 0; r < 8; ++r) {
            float mnew = fmaxf(rmax[r], t[r]);
            scl[r] = __builtin_amdgcn_exp2f((rmax[r] - mnew) * LOG2E);
            rmax[r] = mnew;
            s0[r] = __builtin_amdgcn_exp2f((s0[r] - mnew) * LOG2E);
            s1[r] = __builtin_amdgcn_exp2f((s1[r] - mnew) * LOG2E);
            t[r] = s0[r] + s1[r];
        }
#pragma unroll
        for (int off = 1; off < 16; off <<= 1)
#pragma unroll
            for (int r = 0; r < 8; ++r)
                t[r] += __shfl_xor(t[r], off, 16);
#pragma unroll
        for (int r = 0; r < 8; ++r) {
            rsum[r] = rsum[r] * scl[r] + t[r];
            o0[r] *= scl[r]; o1[r] *= scl[r]; o2[r] *= scl[r]; o3[r] *= scl[r];
        }

        // P (C layout) -> bf16 A fragment via per-wave LDS round trip.
#pragma unroll
        for (int r = 0; r < 8; ++r) {
            int mm = r + 8 * half;
            pW[mm * ASTR + n]      = f2bf(s0[r]);
            pW[mm * ASTR + 16 + n] = f2bf(s1[r]);
        }
        asm volatile("s_wait_dscnt 0" ::: "memory");   // wave-local LDS RAW
        v16bf pa = load_frag_split(&pW32[n * ASTR32 + half * 4]);

        o0 = wmma_bf16(pa, vf[0].v, o0);
        o1 = wmma_bf16(pa, vf[1].v, o1);
        o2 = wmma_bf16(pa, vf[2].v, o2);
        o3 = wmma_bf16(pa, vf[3].v, o3);
    }

    // Epilogue: divide by rowsum, fold 1/32 defactorize mean, atomic reduce.
#pragma unroll
    for (int r = 0; r < 8; ++r) {
        int mm = r + 8 * half;
        int l2 = (qt0 + mm) & 31;
        float sc = (1.0f / 32.0f) / rsum[r];
        atomicAdd(&op[l2 * 64 +      n], o0[r] * sc);
        atomicAdd(&op[l2 * 64 + 16 + n], o1[r] * sc);
        atomicAdd(&op[l2 * 64 + 32 + n], o2[r] * sc);
        atomicAdd(&op[l2 * 64 + 48 + n], o3[r] * sc);
    }
}

// ---------------------------------------------------------------------------
// Kernel 3: output projection. A = o_mean (permuted read), B = ow^T.
// ---------------------------------------------------------------------------
__global__ __launch_bounds__(128) void oproj_kernel(
    const float* __restrict__ oMean, const float* __restrict__ ow,
    const float* __restrict__ obias, float* __restrict__ out)
{
    __shared__ __align__(16) unsigned aT[64 * ASTR32];
    __shared__ __align__(16) unsigned bP[64 * BPSTR];
    const int tid  = threadIdx.x;
    const int lane = tid & 31, w = tid >> 5;
    const int half = lane >> 4, m = lane & 15;
    const int rt = blockIdx.x >> 3, ct = blockIdx.x & 7;

    v8f acc[4];
#pragma unroll
    for (int cg = 0; cg < 4; ++cg) acc[cg] = zero8();

    for (int kk = 0; kk < 512; kk += 32) {
        __syncthreads();
#pragma unroll
        for (int e = 0; e < 8; ++e) {
            int idx = e * 128 + tid;
            int r = idx >> 4, cp = idx & 15;
            int row = rt * 64 + r, col = kk + 2 * cp;   // col even
            // A[row][col] = o_mean[h=col>>6][s2=row&15][l=row>>4][d=col&63]
            const float* src = &oMean[((col >> 6) * 16 + (row & 15)) * 2048 +
                                      (row >> 4) * 64 + (col & 63)];
            aT[r * ASTR32 + cp] = packf2(src[0], src[1]);
        }
#pragma unroll
        for (int e = 0; e < 8; ++e) {
            int idx = e * 128 + tid;
            int kp = idx & 15, c = idx >> 4;
            const float* wr = &ow[(ct * 64 + c) * 512 + kk + 2 * kp];
            bP[c * BPSTR + kp] = packf2(wr[0], wr[1]);
        }
        __syncthreads();

        v16bf a = load_frag_split(&aT[(w * 16 + m) * ASTR32 + half * 4]);
#pragma unroll
        for (int cg = 0; cg < 4; ++cg) {
            v16bf b = load_frag_split(&bP[(cg * 16 + m) * BPSTR + half * 4]);
            acc[cg] = wmma_bf16(a, b, acc[cg]);
        }
    }
#pragma unroll
    for (int cg = 0; cg < 4; ++cg)
#pragma unroll
        for (int r = 0; r < 8; ++r) {
            int row = rt * 64 + w * 16 + r + 8 * half;
            int col = ct * 64 + cg * 16 + m;
            out[row * 512 + col] = acc[cg][r] + obias[col];
        }
}

__global__ void zero_kernel(float* __restrict__ p, int nTotal) {
    int i = blockIdx.x * blockDim.x + threadIdx.x;
    if (i < nTotal) p[i] = 0.0f;
}

// ---------------------------------------------------------------------------
extern "C" void kernel_launch(void* const* d_in, const int* in_sizes, int n_in,
                              void* d_out, int out_size, void* d_ws, size_t ws_size,
                              hipStream_t stream) {
    (void)in_sizes; (void)n_in; (void)out_size; (void)ws_size;
    const float* x    = (const float*)d_in[0];
    const float* qw   = (const float*)d_in[1];
    const float* qb   = (const float*)d_in[2];
    const float* kw   = (const float*)d_in[3];
    const float* kb   = (const float*)d_in[4];
    const float* vw   = (const float*)d_in[5];
    const float* vb   = (const float*)d_in[6];
    const float* ow   = (const float*)d_in[7];
    const float* ob   = (const float*)d_in[8];
    const float* cosT = (const float*)d_in[9];
    const float* sinT = (const float*)d_in[10];

    float* ws = (float*)d_ws;
    float* qB = ws;                    // 512*512 f32
    float* kB = ws + 262144;
    float* vB = ws + 2 * 262144;
    float* oM = ws + 3 * 262144;       // [128][32][64] f32 accumulator

    zero_kernel<<<(262144 + 255) / 256, 256, 0, stream>>>(oM, 262144);
    qkv_gemm_kernel<<<192, 128, 0, stream>>>(x, qw, qb, kw, kb, vw, vb, qB, kB, vB);
    attn_kernel<<<128 * 16, 128, 0, stream>>>(qB, kB, vB, cosT, sinT, oM);
    oproj_kernel<<<64, 128, 0, stream>>>(oM, ow, ob, (float*)d_out);
}
